// LstmCrf_11441792876654
// MI455X (gfx1250) — matrix operand order repaired and checked by reading, not compiled, not run on hardware
//
#include <hip/hip_runtime.h>
#include <hip/hip_bf16.h>

// BiLSTM-CRF for MI455X (gfx1250, wave32, WMMA + async LDS copies).
// B=128, T=512, V=50000, E=100, U=128, 4U=512, K=32.

typedef __attribute__((ext_vector_type(16))) _Float16 v16h;
typedef __attribute__((ext_vector_type(8)))  _Float16 v8h;
typedef __attribute__((ext_vector_type(8)))  float    v8f;
typedef int v4i_g __attribute__((vector_size(16)));   // matches builtin proto

#define Bsz 128
#define Tsz 512
#define Esz 100
#define Usz 128
#define Gsz 512   // 4U
#define Ksz 32
#define KPAD 256  // padded K for [x|h] combined operand
#define ROWP 264  // padded row stride (halves) to avoid LDS bank aliasing

#if __has_builtin(__builtin_amdgcn_global_load_async_to_lds_b128) && \
    __has_builtin(__builtin_amdgcn_s_wait_asynccnt)
#define HAVE_ASYNC_LDS 1
#endif

#ifdef HAVE_ASYNC_LDS
__device__ __forceinline__ void async_copy16(const void* g, void* l) {
  __builtin_amdgcn_global_load_async_to_lds_b128(
      (__attribute__((address_space(1))) v4i_g*)g,
      (__attribute__((address_space(3))) v4i_g*)l, 0, 0);
}
#endif

union V16U { v16h v; v8h h[2]; };

// A-fragment (16-bit A 16x32, ISA 7.12.2): lane L holds M=L%16;
// halves 0..7 <- K = 32*ks + 8*(L/16); halves 8..15 <- K = 32*ks + 16 + 8*(L/16)
__device__ __forceinline__ v16h ld_afrag(const _Float16* rowbase, int ks, int hs) {
  V16U u;
  u.h[0] = *(const v8h*)(rowbase + 32 * ks + 8 * hs);
  u.h[1] = *(const v8h*)(rowbase + 32 * ks + 16 + 8 * hs);
  return u.v;
}

// B-fragment (16-bit B 32x16): lane L holds column N=L%16;
// halves 0..7 <- K = 32*ks + 8*(L/16); halves 8..15 <- K = 32*ks + 16 + 8*(L/16)
__device__ __forceinline__ v16h ld_bfrag(const _Float16* rowbase, int ks, int hs) {
  V16U u;
  u.h[0] = *(const v8h*)(rowbase + 32 * ks + 8 * hs);
  u.h[1] = *(const v8h*)(rowbase + 32 * ks + 16 + 8 * hs);
  return u.v;
}

__device__ __forceinline__ v8f wmma_f16(v16h a, v16h b, v8f c) {
  return __builtin_amdgcn_wmma_f32_16x16x32_f16(false, a, false, b, (short)0, c,
                                                false, false);
}

__device__ __forceinline__ float sigm_(float x) { return 1.0f / (1.0f + __expf(-x)); }
__device__ __forceinline__ float tanh_(float x) { return 2.0f * sigm_(2.0f * x) - 1.0f; }

// ---------------------------------------------------------------------------
// Stage 0: transpose + f16-convert weights into workspace.
//   wT[dir][n][k], n=0..511, k=0..255:  k<100 -> Wk[k][n]; 100..127 -> 0;
//                                       128..255 -> Wr[k-128][n]
//   crfT[n][k],   n=0..31,  k=0..255:  crf_kernel[k][n]
// ---------------------------------------------------------------------------
__global__ void setup_weights_kernel(const float* __restrict__ Wk_f,
                                     const float* __restrict__ Wr_f,
                                     const float* __restrict__ Wk_b,
                                     const float* __restrict__ Wr_b,
                                     const float* __restrict__ crf,
                                     _Float16* __restrict__ wfT,
                                     _Float16* __restrict__ wbT,
                                     _Float16* __restrict__ crfT) {
  int idx = blockIdx.x * 256 + threadIdx.x;
  const int NW = Gsz * KPAD;  // 512*256
  if (idx < 2 * NW) {
    int d = idx / NW, r = idx % NW;
    int n = r >> 8, k = r & 255;
    const float* Wk = d ? Wk_b : Wk_f;
    const float* Wr = d ? Wr_b : Wr_f;
    float v = 0.0f;
    if (k < Esz)        v = Wk[k * Gsz + n];
    else if (k >= 128)  v = Wr[(k - 128) * Gsz + n];
    (d ? wbT : wfT)[n * KPAD + k] = (_Float16)v;
  } else if (idx < 2 * NW + Ksz * KPAD) {
    int r = idx - 2 * NW;
    int n = r >> 8, k = r & 255;
    crfT[n * KPAD + k] = (_Float16)crf[k * Ksz + n];
  }
}

// ---------------------------------------------------------------------------
// Stage 1: both LSTM directions. 16 blocks: dir = blockIdx/8, batch-tile = %8.
// W^T (512x256 f16, padded rows) resident in LDS (~270KB of 320KB WGP LDS),
// staged with GLOBAL_LOAD_ASYNC_TO_LDS_B128 when available. Per step, the
// next timestep's embedding rows are async-copied into an f32 LDS staging
// buffer while the WMMA chain for the current step runs; s_wait_asynccnt 0
// + block barrier publish them before the f32->f16 A-tile conversion.
// Wave w owns hidden units u in [16w,16w+16): its 4 accumulators are gate
// columns u, u+128, u+256, u+384, so gates + cell state stay in registers.
// ---------------------------------------------------------------------------
__global__ __launch_bounds__(256, 1) void bilstm_kernel(
    const int* __restrict__ tokens, const float* __restrict__ emb,
    const _Float16* __restrict__ wfT, const _Float16* __restrict__ wbT,
    const float* __restrict__ b_f, const float* __restrict__ b_b,
    _Float16* __restrict__ hF, _Float16* __restrict__ hB) {
  __shared__ _Float16 Wlds[Gsz * ROWP];   // 512*264*2 = 270336 B
  __shared__ _Float16 Alds[16 * ROWP];    // 16*264*2  =   8448 B
  __shared__ float    Xstg[16 * 128];     // 16*128*4  =   8192 B (async target)

  const int tid  = threadIdx.x;
  const int lane = tid & 31, wave = tid >> 5;
  const int nl = lane & 15, hs = lane >> 4;
  const int u0 = wave * 16;
  const int u  = u0 + nl;

  const int dir = blockIdx.x >> 3;          // 0 = fwd, 1 = bwd
  const int blk = blockIdx.x & 7;           // batch tile
  const _Float16* wT   = dir ? wbT : wfT;
  const float*    bias = dir ? b_b : b_f;
  _Float16*       hOut = dir ? hB : hF;

  // ---- stage W^T into padded LDS rows ----
#ifdef HAVE_ASYNC_LDS
  for (int s = tid; s < Gsz * KPAD / 8; s += 256) {
    int fl = s << 3;
    int n = fl >> 8, k = fl & 255;
    async_copy16(wT + fl, &Wlds[n * ROWP + k]);
  }
  __builtin_amdgcn_s_wait_asynccnt(0);
#else
  for (int s = tid; s < Gsz * KPAD / 8; s += 256) {
    int fl = s << 3;
    int n = fl >> 8, k = fl & 255;
    *(v8h*)(&Wlds[n * ROWP + k]) = *(const v8h*)(wT + fl);
  }
#endif

  // h_0 = 0 in A tile; zero the Xstg pad columns (k in [100,128) stay 0)
  for (int s = tid; s < 16 * Usz; s += 256) {
    int m = s >> 7, k = s & 127;
    Alds[m * ROWP + 128 + k] = (_Float16)0.0f;
    Xstg[s] = 0.0f;
  }
  // x_0 gather (embedding -> f16, pad 100..127 with 0)
  {
    int t0 = dir ? (Tsz - 1) : 0;
    for (int s = tid; s < 16 * 128; s += 256) {
      int m = s >> 7, k = s & 127;
      int tok = tokens[(blk * 16 + m) * Tsz + t0];
      float v = (k < Esz) ? emb[tok * Esz + k] : 0.0f;
      Alds[m * ROWP + k] = (_Float16)v;
    }
  }

  const float bi = bias[u],       bfv = bias[128 + u];
  const float bg = bias[256 + u], bo  = bias[384 + u];
  float cst[8];
#pragma unroll
  for (int r = 0; r < 8; ++r) cst[r] = 0.0f;

  const _Float16* arow = &Alds[nl * ROWP];
  const _Float16* bi_row = &Wlds[(0 * 128 + u) * ROWP];
  const _Float16* bf_row = &Wlds[(1 * 128 + u) * ROWP];
  const _Float16* bg_row = &Wlds[(2 * 128 + u) * ROWP];
  const _Float16* bo_row = &Wlds[(3 * 128 + u) * ROWP];

  for (int t = 0; t < Tsz; ++t) {
    __syncthreads();  // A tile (x_t, h_{t-1}) ready; Xstg free for reuse

#ifdef HAVE_ASYNC_LDS
    // kick off async gather of x_{t+1} (25 x b128 chunks per row); it
    // overlaps the WMMA chain below.
    if (t + 1 < Tsz) {
      int tn = dir ? (Tsz - 2 - t) : (t + 1);
      for (int s = tid; s < 16 * 25; s += 256) {
        int m = s / 25, c = s % 25;
        int tok = tokens[(blk * 16 + m) * Tsz + tn];
        async_copy16(emb + (size_t)tok * Esz + c * 4, &Xstg[m * 128 + c * 4]);
      }
    }
#endif

    v8f zi = {bi, bi, bi, bi, bi, bi, bi, bi};
    v8f zf = {bfv, bfv, bfv, bfv, bfv, bfv, bfv, bfv};
    v8f zg = {bg, bg, bg, bg, bg, bg, bg, bg};
    v8f zo = {bo, bo, bo, bo, bo, bo, bo, bo};
#pragma unroll
    for (int ks = 0; ks < 8; ++ks) {          // K = 256 / 32
      v16h a = ld_afrag(arow, ks, hs);
      zi = wmma_f16(a, ld_bfrag(bi_row, ks, hs), zi);
      zf = wmma_f16(a, ld_bfrag(bf_row, ks, hs), zf);
      zg = wmma_f16(a, ld_bfrag(bg_row, ks, hs), zg);
      zo = wmma_f16(a, ld_bfrag(bo_row, ks, hs), zo);
    }

#ifdef HAVE_ASYNC_LDS
    if (t + 1 < Tsz) __builtin_amdgcn_s_wait_asynccnt(0);
#endif
    __syncthreads();  // all waves done reading A; all async x_{t+1} in LDS

    int t_act = dir ? (Tsz - 1 - t) : t;
#pragma unroll
    for (int r = 0; r < 8; ++r) {  // C layout: M = r + 8*(lane/16), N = lane%16
      float cn = sigm_(zf[r]) * cst[r] + sigm_(zi[r]) * tanh_(zg[r]);
      cst[r] = cn;
      float h = sigm_(zo[r]) * tanh_(cn);
      int m = r + 8 * hs;
      Alds[m * ROWP + 128 + u] = (_Float16)h;
      hOut[((size_t)(blk * 16 + m) * Tsz + t_act) * Usz + u] = (_Float16)h;
    }

    if (t + 1 < Tsz) {  // publish x_{t+1} into the f16 A tile
#ifdef HAVE_ASYNC_LDS
      for (int s = tid; s < 16 * 128; s += 256) {
        int m = s >> 7, k = s & 127;
        Alds[m * ROWP + k] = (_Float16)Xstg[s];
      }
#else
      int tn = dir ? (Tsz - 2 - t) : (t + 1);
      for (int s = tid; s < 16 * 128; s += 256) {
        int m = s >> 7, k = s & 127;
        int tok = tokens[(blk * 16 + m) * Tsz + tn];
        float v = (k < Esz) ? emb[tok * Esz + k] : 0.0f;
        Alds[m * ROWP + k] = (_Float16)v;
      }
#endif
    }
  }
}

// ---------------------------------------------------------------------------
// Stage 2: emissions em = [hF|hB] @ crf_kernel + bias. M=65536, K=256, N=32.
// 1024 blocks * 8 waves; wave = (M-subtile 0..3, N-tile 0..1). Fragments read
// straight from global (crfT is 16KB, hot in L2).
// ---------------------------------------------------------------------------
__global__ __launch_bounds__(256, 4) void emissions_kernel(
    const _Float16* __restrict__ hF, const _Float16* __restrict__ hB,
    const _Float16* __restrict__ crfT, const float* __restrict__ cbias,
    float* __restrict__ em) {
  int lane = threadIdx.x & 31, wave = threadIdx.x >> 5;
  int nl = lane & 15, hs = lane >> 4;
  int tile  = blockIdx.x * 4 + (wave & 3);
  int ntile = wave >> 2;
  int bt = tile * 16 + nl;  // A row for this lane (M = lane%16)

  float bv = cbias[ntile * 16 + nl];
  v8f acc = {bv, bv, bv, bv, bv, bv, bv, bv};
  const _Float16* arow_f = hF + (size_t)bt * Usz;
  const _Float16* arow_b = hB + (size_t)bt * Usz;
  const _Float16* brow   = crfT + (size_t)(ntile * 16 + nl) * KPAD;
#pragma unroll
  for (int ks = 0; ks < 8; ++ks) {
    int kk = 32 * ks;
    const _Float16* ap = (kk < 128) ? (arow_f + kk) : (arow_b + (kk - 128));
    v16h a = ld_afrag(ap, 0, hs);
    v16h b = ld_bfrag(brow + kk, 0, hs);
    acc = wmma_f16(a, b, acc);
  }
#pragma unroll
  for (int r = 0; r < 8; ++r) {
    int m = r + 8 * hs;
    em[(size_t)(tile * 16 + m) * Ksz + ntile * 16 + nl] = acc[r];
  }
}

// ---------------------------------------------------------------------------
// Stage 3: CRF logZ. One wave32 per batch row; lane j carries alpha[j].
// trans staged in LDS with 33-stride (conflict-free column reads).
// ---------------------------------------------------------------------------
__global__ __launch_bounds__(128, 1) void crf_logz_kernel(
    const float* __restrict__ em, const float* __restrict__ trans,
    float* __restrict__ out) {
  __shared__ float tr[32 * 33];
  int tid = threadIdx.x, lane = tid & 31, wave = tid >> 5;
  for (int s = tid; s < 1024; s += 128) tr[(s >> 5) * 33 + (s & 31)] = trans[s];
  __syncthreads();

  int b = blockIdx.x * 4 + wave;
  const float* eb = em + (size_t)b * Tsz * Ksz;
  float alpha = eb[lane];
  for (int t = 1; t < Tsz; ++t) {
    float et = eb[t * Ksz + lane];
    float m = -3.0e38f;
    for (int i = 0; i < 32; ++i) {
      float v = __shfl(alpha, i, 32) + tr[i * 33 + lane];
      m = fmaxf(m, v);
    }
    float s = 0.0f;
    for (int i = 0; i < 32; ++i) {
      float v = __shfl(alpha, i, 32) + tr[i * 33 + lane];
      s += __expf(v - m);
    }
    alpha = m + __logf(s) + et;
  }
  float m = alpha;
  for (int off = 16; off; off >>= 1) m = fmaxf(m, __shfl_xor(m, off, 32));
  float s = __expf(alpha - m);
  for (int off = 16; off; off >>= 1) s += __shfl_xor(s, off, 32);
  if (lane == 0) out[b] = m + __logf(s);
}

// ---------------------------------------------------------------------------
extern "C" void kernel_launch(void* const* d_in, const int* in_sizes, int n_in,
                              void* d_out, int out_size, void* d_ws, size_t ws_size,
                              hipStream_t stream) {
  const int*   tokens = (const int*)d_in[0];
  const float* emb    = (const float*)d_in[1];
  const float* Wk_f   = (const float*)d_in[2];
  const float* Wr_f   = (const float*)d_in[3];
  const float* b_f    = (const float*)d_in[4];
  const float* Wk_b   = (const float*)d_in[5];
  const float* Wr_b   = (const float*)d_in[6];
  const float* b_b    = (const float*)d_in[7];
  const float* crf    = (const float*)d_in[8];
  const float* cbias  = (const float*)d_in[9];
  const float* trans  = (const float*)d_in[10];
  float* out = (float*)d_out;

  char* ws = (char*)d_ws;
  const size_t SZ_W   = (size_t)Gsz * KPAD * 2;            // 262144
  const size_t SZ_CRF = (size_t)Ksz * KPAD * 2;            // 16384
  const size_t SZ_H   = (size_t)Bsz * Tsz * Usz * 2;       // 16777216
  _Float16* wfT  = (_Float16*)(ws);
  _Float16* wbT  = (_Float16*)(ws + SZ_W);
  _Float16* crfT = (_Float16*)(ws + 2 * SZ_W);
  _Float16* hF   = (_Float16*)(ws + 2 * SZ_W + SZ_CRF);
  _Float16* hB   = (_Float16*)(ws + 2 * SZ_W + SZ_CRF + SZ_H);
  float*    em   = (float*)  (ws + 2 * SZ_W + SZ_CRF + 2 * SZ_H);

  int setup_elems = 2 * Gsz * KPAD + Ksz * KPAD;  // 270336
  setup_weights_kernel<<<(setup_elems + 255) / 256, 256, 0, stream>>>(
      Wk_f, Wr_f, Wk_b, Wr_b, crf, wfT, wbT, crfT);

  bilstm_kernel<<<16, 256, 0, stream>>>(tokens, emb, wfT, wbT, b_f, b_b, hF, hB);

  emissions_kernel<<<(Bsz * Tsz / 16) / 4, 256, 0, stream>>>(hF, hB, crfT, cbias, em);

  crf_logz_kernel<<<Bsz / 4, 128, 0, stream>>>(em, trans, out);
}
